// MonaiSwin_7438883357270
// MI455X (gfx1250) — compile-verified, matrix operand order checked by reading, not run on hardware
//
#include <hip/hip_runtime.h>
#include <hip/hip_bf16.h>
#include <math.h>

typedef __attribute__((ext_vector_type(16))) _Float16 v16h;
typedef __attribute__((ext_vector_type(8)))  _Float16 v8h;
typedef __attribute__((ext_vector_type(8)))  float    v8f;

#define LN_EPS 1e-5f

// ------------------------------------------------------------------
// Patch embed: Conv2d(3,24,k=2,s=2) == patchify(12) @ W^T + b
// x: (16,3,224,224) NCHW fp32 -> tokens (16*112*112, 24) fp32
// ------------------------------------------------------------------
__global__ void k_patch_embed(const float* __restrict__ x, const float* __restrict__ w,
                              const float* __restrict__ bias, float* __restrict__ out)
{
  size_t idx = (size_t)blockIdx.x * blockDim.x + threadIdx.x;
  const size_t total = (size_t)16 * 112 * 112 * 24;
  if (idx >= total) return;
  int co = (int)(idx % 24);
  size_t tok = idx / 24;
  int j = (int)(tok % 112); size_t r = tok / 112;
  int i = (int)(r % 112);   int b = (int)(r / 112);
  float acc = bias[co];
#pragma unroll
  for (int ci = 0; ci < 3; ++ci)
#pragma unroll
    for (int kh = 0; kh < 2; ++kh)
#pragma unroll
      for (int kw = 0; kw < 2; ++kw)
        acc += x[(((size_t)b * 3 + ci) * 224 + 2 * i + kh) * 224 + 2 * j + kw] *
               w[((co * 3 + ci) * 2 + kh) * 2 + kw];
  out[idx] = acc;
}

// ------------------------------------------------------------------
// Weight pack: fp32 [N,K] row-major -> f16 [Nr,Kp] zero padded
// (Nr rounded to 32 so the GEMM can run 2 N-tiles per wave guard-free)
// ------------------------------------------------------------------
__global__ void k_pack_w(const float* __restrict__ w, _Float16* __restrict__ out,
                         int N, int K, int Nr, int Kp)
{
  size_t idx = (size_t)blockIdx.x * blockDim.x + threadIdx.x;
  size_t total = (size_t)Nr * Kp;
  if (idx >= total) return;
  int k = (int)(idx % Kp);
  int n = (int)(idx / Kp);
  float v = (n < N && k < K) ? w[(size_t)n * K + k] : 0.0f;
  out[idx] = (_Float16)v;
}

// ------------------------------------------------------------------
// LayerNorm fused with (shifted) window partition; f16 padded output.
// 8 tokens per 256-thread block, one wave32 per token.
// ------------------------------------------------------------------
__global__ void k_ln_part(const float* __restrict__ x, const float* __restrict__ g,
                          const float* __restrict__ be, _Float16* __restrict__ xw,
                          int Ntok, int H, int W, int C, int Kp, int shift)
{
  int tok  = blockIdx.x * 8 + (threadIdx.x >> 5);
  if (tok >= Ntok) return;
  int lane = threadIdx.x & 31;
  int j = tok % W; int r = tok / W;
  int i = r % H;   int b = r / H;
  const float* row = x + (size_t)tok * C;
  float s = 0.f, s2 = 0.f;
  for (int c = lane; c < C; c += 32) { float v = row[c]; s += v; s2 += v * v; }
#pragma unroll
  for (int off = 16; off; off >>= 1) {
    s  += __shfl_xor(s,  off, 32);
    s2 += __shfl_xor(s2, off, 32);
  }
  float mean = s / C;
  float rstd = rsqrtf(s2 / C - mean * mean + LN_EPS);
  // destination slot in the rolled (shifted) image: xsh[si] = x[(si+shift)%H]
  int si = i - shift; if (si < 0) si += H;
  int sj = j - shift; if (sj < 0) sj += W;
  int win = (b * (H / 7) + si / 7) * (W / 7) + sj / 7;
  _Float16* dst = xw + ((size_t)win * 49 + (si % 7) * 7 + (sj % 7)) * Kp;
  for (int c = lane; c < Kp; c += 32) {
    _Float16 hv = (_Float16)0.f;
    if (c < C) hv = (_Float16)((row[c] - mean) * rstd * g[c] + be[c]);
    dst[c] = hv;
  }
}

// Plain LayerNorm -> f16 padded (gamma/beta may be null => no affine).
__global__ void k_ln(const float* __restrict__ x, const float* __restrict__ g,
                     const float* __restrict__ be, _Float16* __restrict__ out,
                     int Ntok, int C, int Kp)
{
  int tok  = blockIdx.x * 8 + (threadIdx.x >> 5);
  if (tok >= Ntok) return;
  int lane = threadIdx.x & 31;
  const float* row = x + (size_t)tok * C;
  float s = 0.f, s2 = 0.f;
  for (int c = lane; c < C; c += 32) { float v = row[c]; s += v; s2 += v * v; }
#pragma unroll
  for (int off = 16; off; off >>= 1) {
    s  += __shfl_xor(s,  off, 32);
    s2 += __shfl_xor(s2, off, 32);
  }
  float mean = s / C;
  float rstd = rsqrtf(s2 / C - mean * mean + LN_EPS);
  _Float16* dst = out + (size_t)tok * Kp;
  for (int c = lane; c < Kp; c += 32) {
    _Float16 hv = (_Float16)0.f;
    if (c < C) {
      float v = (row[c] - mean) * rstd;
      if (g) v = v * g[c] + be[c];
      hv = (_Float16)v;
    }
    dst[c] = hv;
  }
}

// ------------------------------------------------------------------
// WMMA GEMM: out[M,N](f32) = Ah[M,Kp](f16) @ Bh[Nr,Kp](f16)^T (+bias)(+resid)
// One wave per 16x32 macro-tile (2 N-tiles sharing the A fragment):
// per 32-K slab: 6x global_load_b128 + 2x v_wmma, branch-free.
// M must be a multiple of 16 (true for every GEMM in this network);
// Nr is a multiple of 32 (weight pack pads with zero rows).
// ------------------------------------------------------------------
__global__ void k_gemm_wmma(const _Float16* __restrict__ Ah, const _Float16* __restrict__ Bh,
                            const float* __restrict__ bias, const float* __restrict__ resid,
                            float* __restrict__ out, int M, int N, int Kp)
{
  const int lane = threadIdx.x;          // 0..31
  const int col0 = blockIdx.x * 32;
  const int row0 = blockIdx.y * 16;
  const int kb   = (lane & 16) ? 8 : 0;  // ISA 7.12.2 K-half select
  const unsigned aoff = (unsigned)(row0 + (lane & 15)) * (unsigned)Kp + kb;
  const unsigned boff = (unsigned)(col0 + (lane & 15)) * (unsigned)Kp + kb;
  const _Float16* ap  = Ah + aoff;
  const _Float16* bp0 = Bh + boff;
  const _Float16* bp1 = bp0 + (unsigned)16 * (unsigned)Kp;
  v8f acc0 = {}, acc1 = {};
#pragma unroll 4
  for (int k0 = 0; k0 < Kp; k0 += 32) {
    __builtin_prefetch(ap + k0 + 128, 0, 1);
    v8h a0 = *(const v8h*)(ap + k0);
    v8h a1 = *(const v8h*)(ap + k0 + 16);
    v8h p0 = *(const v8h*)(bp0 + k0);
    v8h p1 = *(const v8h*)(bp0 + k0 + 16);
    v8h q0 = *(const v8h*)(bp1 + k0);
    v8h q1 = *(const v8h*)(bp1 + k0 + 16);
    v16h a  = __builtin_shufflevector(a0, a1, 0,1,2,3,4,5,6,7,8,9,10,11,12,13,14,15);
    v16h b0 = __builtin_shufflevector(p0, p1, 0,1,2,3,4,5,6,7,8,9,10,11,12,13,14,15);
    v16h b1 = __builtin_shufflevector(q0, q1, 0,1,2,3,4,5,6,7,8,9,10,11,12,13,14,15);
    acc0 = __builtin_amdgcn_wmma_f32_16x16x32_f16(false, a, false, b0,
                                                  (short)0, acc0, false, false);
    acc1 = __builtin_amdgcn_wmma_f32_16x16x32_f16(false, a, false, b1,
                                                  (short)0, acc1, false, false);
  }
#pragma unroll
  for (int rr = 0; rr < 8; ++rr) {
    int row = row0 + rr + ((lane & 16) ? 8 : 0);
    unsigned rbase = (unsigned)row * (unsigned)N;
    int c0 = col0 + (lane & 15);
    int c1 = c0 + 16;
    if (c0 < N) {
      float v = acc0[rr];
      if (bias)  v += bias[c0];
      if (resid) v += resid[rbase + c0];
      out[rbase + c0] = v;
    }
    if (c1 < N) {
      float v = acc1[rr];
      if (bias)  v += bias[c1];
      if (resid) v += resid[rbase + c1];
      out[rbase + c1] = v;
    }
  }
}

// ------------------------------------------------------------------
// Fused window attention (head_dim == 8 everywhere):
// scores (49x49, rel-pos bias + shifted-window mask inline), softmax,
// P@V in LDS. One 64-thread block per (window, head). f16 padded out.
// ------------------------------------------------------------------
__device__ __forceinline__ int reg3(int s, int L) {
  return (s < L - 7) ? 0 : (s < L - 3) ? 1 : 2;
}

__global__ void k_attn(const float* __restrict__ qkv, const float* __restrict__ rpb,
                       _Float16* __restrict__ out, int nWh, int nWw, int nheads, int C,
                       int Cp, int H, int W, int shift)
{
  const int win  = blockIdx.x;     // b * nWh * nWw
  const int head = blockIdx.y;
  const int t    = threadIdx.x;    // 0..63
  __shared__ float sq[49 * 8], sk[49 * 8], sv[49 * 8], sS[49 * 49];
  const int C3 = 3 * C;
  const size_t base = (size_t)win * 49 * C3 + head * 8;
  const float scale = 0.35355339059327379f;   // 8^-0.5
  for (int idx = t; idx < 392; idx += 64) {
    int m = idx >> 3, d = idx & 7;
    const float* p = qkv + base + (size_t)m * C3 + d;
    sq[idx] = p[0] * scale;
    sk[idx] = p[C];
    sv[idx] = p[2 * C];
  }
  __syncthreads();
  int wrem = win % (nWh * nWw);
  int wi = wrem / nWw, wj = wrem % nWw;
  for (int p = t; p < 2401; p += 64) {
    int m = p / 49, nn = p % 49;
    float acc = 0.f;
#pragma unroll
    for (int d = 0; d < 8; ++d) acc += sq[m * 8 + d] * sk[nn * 8 + d];
    int mi = m / 7, mj = m % 7, ni = nn / 7, nj = nn % 7;
    acc += rpb[((mi - ni + 6) * 13 + (mj - nj + 6)) * nheads + head];
    if (shift) {
      int rm = reg3(wi * 7 + mi, H) * 3 + reg3(wj * 7 + mj, W);
      int rn = reg3(wi * 7 + ni, H) * 3 + reg3(wj * 7 + nj, W);
      if (rm != rn) acc -= 100.f;
    }
    sS[p] = acc;
  }
  __syncthreads();
  if (t < 49) {
    float mx = -1e30f;
    for (int nn = 0; nn < 49; ++nn) mx = fmaxf(mx, sS[t * 49 + nn]);
    float sum = 0.f;
    for (int nn = 0; nn < 49; ++nn) {
      float e = __expf(sS[t * 49 + nn] - mx);
      sS[t * 49 + nn] = e; sum += e;
    }
    float inv = 1.f / sum;
    for (int nn = 0; nn < 49; ++nn) sS[t * 49 + nn] *= inv;
  }
  __syncthreads();
  for (int idx = t; idx < 392; idx += 64) {
    int m = idx >> 3, d = idx & 7;
    float acc = 0.f;
    for (int nn = 0; nn < 49; ++nn) acc += sS[m * 49 + nn] * sv[nn * 8 + d];
    out[((size_t)win * 49 + m) * Cp + head * 8 + d] = (_Float16)acc;
  }
  // zero the K-pad columns once (head 0 covers its window's rows)
  if (head == 0 && Cp > C) {
    int pw = Cp - C;
    for (int idx = t; idx < 49 * pw; idx += 64) {
      int m = idx / pw, c = C + idx % pw;
      out[((size_t)win * 49 + m) * Cp + c] = (_Float16)0.f;
    }
  }
}

// Window-reverse (+un-shift) fused with residual add: out = skip + scatter(wbuf)
__global__ void k_winrev_add(const float* __restrict__ skip, const float* __restrict__ wbuf,
                             float* __restrict__ out, int B, int H, int W, int C, int shift)
{
  size_t total = (size_t)B * H * W * C;
  size_t idx = (size_t)blockIdx.x * blockDim.x + threadIdx.x;
  if (idx >= total) return;
  int c = (int)(idx % C); size_t tok = idx / C;
  int j = (int)(tok % W); size_t r = tok / W;
  int i = (int)(r % H);   int b = (int)(r / H);
  int si = i - shift; if (si < 0) si += H;
  int sj = j - shift; if (sj < 0) sj += W;
  int win = (b * (H / 7) + si / 7) * (W / 7) + sj / 7;
  out[idx] = skip[idx] + wbuf[((size_t)win * 49 + (si % 7) * 7 + (sj % 7)) * C + c];
}

// GELU(exact) fp32 -> f16 (width already a multiple of 32, no pad needed)
__global__ void k_gelu_h(const float* __restrict__ x, _Float16* __restrict__ out, size_t n)
{
  size_t i = (size_t)blockIdx.x * blockDim.x + threadIdx.x;
  if (i < n) {
    float v = x[i];
    out[i] = (_Float16)(0.5f * v * (1.f + erff(v * 0.70710678118654752f)));
  }
}

// Patch-merge concat: (B,H,W,C) -> (B,H/2,W/2,4C), order [(0,0),(1,0),(0,1),(1,1)]
__global__ void k_merge(const float* __restrict__ x, float* __restrict__ out,
                        int B, int H, int W, int C)
{
  size_t total = (size_t)B * (H / 2) * (W / 2) * 4 * C;
  size_t idx = (size_t)blockIdx.x * blockDim.x + threadIdx.x;
  if (idx >= total) return;
  int c4 = (int)(idx % (4 * C)); size_t tok = idx / (4 * C);
  int q = c4 / C, c = c4 % C;
  int j2 = (int)(tok % (W / 2)); size_t r = tok / (W / 2);
  int i2 = (int)(r % (H / 2));   int b = (int)(r / (H / 2));
  int di = q & 1, dj = q >> 1;
  out[idx] = x[(((size_t)b * H + 2 * i2 + di) * W + 2 * j2 + dj) * C + c];
}

// mean over T tokens, f16 in -> f16 out (feeds head GEMM)
__global__ void k_mean_h(const _Float16* __restrict__ x, _Float16* __restrict__ feat,
                         int B, int T, int C)
{
  int idx = blockIdx.x * blockDim.x + threadIdx.x;
  if (idx >= B * C) return;
  int c = idx % C, b = idx / C;
  float s = 0.f;
  for (int t = 0; t < T; ++t) s += (float)x[((size_t)b * T + t) * C + c];
  feat[idx] = (_Float16)(s / (float)T);
}

// ------------------------------------------------------------------
// Host side
// ------------------------------------------------------------------
struct BlockP {
  const float *n1_g, *n1_b, *qkv_w, *qkv_b, *rpb, *proj_w, *proj_b;
  const float *n2_g, *n2_b, *fc1_w, *fc1_b, *fc2_w, *fc2_b;
};
struct StageP { BlockP blk[2]; const float *ds_g, *ds_b, *red_w; };
struct PackedW { _Float16* ptr; int Nr, Kp; };

static inline int rup(int v, int a) { return (v + a - 1) & ~(a - 1); }

static inline void gemm(const _Float16* Ah, const PackedW& Wp, const float* bias,
                        const float* resid, float* out, int M, int N, hipStream_t s)
{
  dim3 g((unsigned)(Wp.Nr / 32), (unsigned)(M / 16));
  k_gemm_wmma<<<g, dim3(32), 0, s>>>(Ah, Wp.ptr, bias, resid, out, M, N, Wp.Kp);
}

extern "C" void kernel_launch(void* const* d_in, const int* in_sizes, int n_in,
                              void* d_out, int out_size, void* d_ws, size_t ws_size,
                              hipStream_t stream)
{
  (void)n_in; (void)out_size; (void)ws_size;
  auto F = [&](int i) { return (const float*)d_in[i]; };

  const float *x_in, *pe_w, *pe_b, *head_w, *head_b;
  StageP st[4];
  const int X_ELEMS = 16 * 3 * 224 * 224;   // 2408448, unique size

  if (in_sizes[0] == X_ELEMS) {
    // insertion-order flatten: x, pe_w, pe_b, stages..., head_w, head_b
    int i = 0;
    x_in = F(i++); pe_w = F(i++); pe_b = F(i++);
    for (int s = 0; s < 4; ++s) {
      for (int bl = 0; bl < 2; ++bl) {
        BlockP& B = st[s].blk[bl];
        B.n1_g = F(i++); B.n1_b = F(i++); B.qkv_w = F(i++); B.qkv_b = F(i++);
        B.rpb = F(i++);  B.proj_w = F(i++); B.proj_b = F(i++);
        B.n2_g = F(i++); B.n2_b = F(i++);
        B.fc1_w = F(i++); B.fc1_b = F(i++); B.fc2_w = F(i++); B.fc2_b = F(i++);
      }
      st[s].ds_g = F(i++); st[s].ds_b = F(i++); st[s].red_w = F(i++);
    }
    head_w = F(i++); head_b = F(i++);
  } else {
    // jax pytree sorted-key flatten: head_b, head_w, pe_b, pe_w, stages..., x
    int i = 0;
    head_b = F(i++); head_w = F(i++); pe_b = F(i++); pe_w = F(i++);
    for (int s = 0; s < 4; ++s) {
      for (int bl = 0; bl < 2; ++bl) {
        BlockP& B = st[s].blk[bl];
        B.fc1_b = F(i++); B.fc1_w = F(i++); B.fc2_b = F(i++); B.fc2_w = F(i++);
        B.n1_b = F(i++);  B.n1_g = F(i++);  B.n2_b = F(i++);  B.n2_g = F(i++);
        B.proj_b = F(i++); B.proj_w = F(i++); B.qkv_b = F(i++); B.qkv_w = F(i++);
        B.rpb = F(i++);
      }
      st[s].ds_b = F(i++); st[s].ds_g = F(i++); st[s].red_w = F(i++);
    }
    x_in = F(i++);
  }

  // ---- workspace arenas ----
  const size_t NC = (size_t)16 * 112 * 112 * 24;   // 4,816,896
  char* base = (char*)d_ws;
  auto carve = [&](size_t bytes) {
    char* p = base;
    base += (bytes + 255) & ~(size_t)255;
    return p;
  };
  float*    t0  = (float*)   carve(NC * 4);          // residual stream (B,H,W,C) f32
  float*    t2  = (float*)   carve(NC * 4);          // proj-out / merge buffer f32
  float*    q32 = (float*)   carve(NC * 4 * 4);      // qkv f32 & fc1-out f32 (M*4C)
  _Float16* xh  = (_Float16*)carve(NC * 32 / 24 * 2);// LN out f16 padded (max M*32)
  _Float16* hh  = (_Float16*)carve(NC * 4 * 2);      // gelu/attn-out f16 (max M*4C)
  _Float16* wh  = (_Float16*)carve((size_t)2097152 * 2); // packed weights f16

  // ---- pack all weights (f16, zero padded, Nr%32==0) ----
  size_t wcur = 0;
  auto pack = [&](const float* w, int N, int K) {
    PackedW pw;
    pw.Nr = rup(N, 32); pw.Kp = rup(K, 32);
    pw.ptr = wh + wcur;
    size_t total = (size_t)pw.Nr * pw.Kp;
    wcur += total;
    k_pack_w<<<dim3((unsigned)((total + 255) / 256)), dim3(256), 0, stream>>>(
        w, pw.ptr, N, K, pw.Nr, pw.Kp);
    return pw;
  };

  const int B = 16;
  const int Cs[4]  = {24, 48, 96, 192};
  const int HWs[4] = {112, 56, 28, 14};
  const int NHs[4] = {3, 6, 12, 24};

  PackedW Wqkv[4][2], Wproj[4][2], Wfc1[4][2], Wfc2[4][2], Wred[4], Whead;
  for (int s = 0; s < 4; ++s) {
    const int C = Cs[s];
    for (int bl = 0; bl < 2; ++bl) {
      const BlockP& P = st[s].blk[bl];
      Wqkv[s][bl]  = pack(P.qkv_w,  3 * C, C);
      Wproj[s][bl] = pack(P.proj_w, C,     C);
      Wfc1[s][bl]  = pack(P.fc1_w,  4 * C, C);
      Wfc2[s][bl]  = pack(P.fc2_w,  C,     4 * C);
    }
    Wred[s] = pack(st[s].red_w, 2 * C, 4 * C);
  }
  Whead = pack(head_w, 5, 384);

  // ---- patch embed -> t0 ----
  {
    size_t total = (size_t)B * 112 * 112 * 24;
    k_patch_embed<<<dim3((unsigned)((total + 255) / 256)), dim3(256), 0, stream>>>(
        x_in, pe_w, pe_b, t0);
  }

  // ---- stages ----
  for (int s = 0; s < 4; ++s) {
    const int C = Cs[s], H = HWs[s], W = HWs[s], NH = NHs[s];
    const int Cp = rup(C, 32);
    const int Ntok = B * H * W;
    const int nWh = H / 7, nWw = W / 7;
    const size_t elems = (size_t)Ntok * C;

    for (int bl = 0; bl < 2; ++bl) {
      const BlockP& P = st[s].blk[bl];
      const int shift = bl ? 3 : 0;

      // LN1 + shifted window partition -> xh (f16, [Ntok, Cp], window order)
      k_ln_part<<<dim3((unsigned)((Ntok + 7) / 8)), dim3(256), 0, stream>>>(
          t0, P.n1_g, P.n1_b, xh, Ntok, H, W, C, Cp, shift);
      // qkv: (Ntok,Cp)f16 @ Wqkv -> q32 (f32, [Ntok,3C])
      gemm(xh, Wqkv[s][bl], P.qkv_b, nullptr, q32, Ntok, 3 * C, stream);
      // fused window attention -> hh (f16, [Ntok, Cp], window order)
      k_attn<<<dim3((unsigned)(B * nWh * nWw), (unsigned)NH), dim3(64), 0, stream>>>(
          q32, P.rpb, hh, nWh, nWw, NH, C, Cp, H, W, shift);
      // proj -> t2 (f32, window order)
      gemm(hh, Wproj[s][bl], P.proj_b, nullptr, t2, Ntok, C, stream);
      // window reverse + un-shift + residual -> t0
      k_winrev_add<<<dim3((unsigned)((elems + 255) / 256)), dim3(256), 0, stream>>>(
          t0, t2, t0, B, H, W, C, shift);
      // LN2 -> xh (f16 padded)
      k_ln<<<dim3((unsigned)((Ntok + 7) / 8)), dim3(256), 0, stream>>>(
          t0, P.n2_g, P.n2_b, xh, Ntok, C, Cp);
      // fc1 -> q32 (f32, [Ntok,4C]); GELU -> hh (f16); fc2 (+resid t0) -> t0
      gemm(xh, Wfc1[s][bl], P.fc1_b, nullptr, q32, Ntok, 4 * C, stream);
      {
        size_t hn = (size_t)Ntok * 4 * C;
        k_gelu_h<<<dim3((unsigned)((hn + 255) / 256)), dim3(256), 0, stream>>>(
            q32, hh, hn);
      }
      gemm(hh, Wfc2[s][bl], P.fc2_b, t0, t0, Ntok, C, stream);
    }

    // patch merging: concat -> t2 (f32, [Ntok/4, 4C]); LN -> xh (f16); reduce -> t0
    {
      size_t total = (size_t)Ntok * C;
      k_merge<<<dim3((unsigned)((total + 255) / 256)), dim3(256), 0, stream>>>(
          t0, t2, B, H, W, C);
      k_ln<<<dim3((unsigned)((Ntok / 4 + 7) / 8)), dim3(256), 0, stream>>>(
          t2, st[s].ds_g, st[s].ds_b, xh, Ntok / 4, 4 * C, 4 * C);
      gemm(xh, Wred[s], nullptr, nullptr, t0, Ntok / 4, 2 * C, stream);
    }
  }

  // ---- final: LN(no affine) -> xh f16; mean over 49 tokens -> hh; head -> d_out
  const int Tf = 7 * 7, Cf = 384;
  k_ln<<<dim3((unsigned)((B * Tf + 7) / 8)), dim3(256), 0, stream>>>(
      t0, nullptr, nullptr, xh, B * Tf, Cf, Cf);
  k_mean_h<<<dim3((unsigned)((B * Cf + 255) / 256)), dim3(256), 0, stream>>>(
      xh, hh, B, Tf, Cf);
  gemm(hh, Whead, head_b, nullptr, (float*)d_out, B, 5, stream);
}
